// SetToSeqAgent_42030549958825
// MI455X (gfx1250) — matrix operand order repaired and checked by reading, not compiled, not run on hardware
//
#include <hip/hip_runtime.h>
#include <hip/hip_bf16.h>
#include <math.h>

typedef __attribute__((ext_vector_type(16))) _Float16 v16h;
typedef __attribute__((ext_vector_type(8)))  float    v8f;

#define B_   32
#define L_   128
#define DV_  64
#define D_   512
#define H_   8
#define HD_  64
#define FF_  2048
#define NE_  4
#define ND_  4
#define BL_  (B_ * L_)

#define WMMA_F16(a, b, c) \
  __builtin_amdgcn_wmma_f32_16x16x32_f16(false, (a), false, (b), (short)0, (c), false, false)

// ---------------------------------------------------------------------------
// WMMA fragment loaders (wave32, 16x16x32 f16 shapes per CDNA5 ISA)
// A 16x32: lanes 0-15 -> rows 0-15, elems 0-15 = K klo..klo+15
//          (klo = 0 for lanes<16, 16 for lanes>=16)
// B 32x16: lanes 0-15 -> cols, K 0-15; lanes 16-31 -> cols, K 16-31
// C/D 16x16: vgpr r, lanes 0-15 -> row r, lanes 16-31 -> row r+8; col = lane&15
// ---------------------------------------------------------------------------
__device__ __forceinline__ v16h load_a32(const float* __restrict__ row, int k, int lane) {
  const int klo = (lane & 16) ? 16 : 0;
  const float4 f0 = *(const float4*)(row + k + klo);
  const float4 f1 = *(const float4*)(row + k + klo + 4);
  const float4 f2 = *(const float4*)(row + k + klo + 8);
  const float4 f3 = *(const float4*)(row + k + klo + 12);
  v16h a;
  a[0]  = (_Float16)f0.x; a[1]  = (_Float16)f0.y; a[2]  = (_Float16)f0.z; a[3]  = (_Float16)f0.w;
  a[4]  = (_Float16)f1.x; a[5]  = (_Float16)f1.y; a[6]  = (_Float16)f1.z; a[7]  = (_Float16)f1.w;
  a[8]  = (_Float16)f2.x; a[9]  = (_Float16)f2.y; a[10] = (_Float16)f2.z; a[11] = (_Float16)f2.w;
  a[12] = (_Float16)f3.x; a[13] = (_Float16)f3.y; a[14] = (_Float16)f3.z; a[15] = (_Float16)f3.w;
  return a;
}

__device__ __forceinline__ v16h load_b32(const float* __restrict__ row, int k, int lane) {
  const int koff = (lane & 16) ? 16 : 0;
  const float4 f0 = *(const float4*)(row + k + koff);
  const float4 f1 = *(const float4*)(row + k + koff + 4);
  const float4 f2 = *(const float4*)(row + k + koff + 8);
  const float4 f3 = *(const float4*)(row + k + koff + 12);
  v16h b;
  b[0]  = (_Float16)f0.x; b[1]  = (_Float16)f0.y; b[2]  = (_Float16)f0.z; b[3]  = (_Float16)f0.w;
  b[4]  = (_Float16)f1.x; b[5]  = (_Float16)f1.y; b[6]  = (_Float16)f1.z; b[7]  = (_Float16)f1.w;
  b[8]  = (_Float16)f2.x; b[9]  = (_Float16)f2.y; b[10] = (_Float16)f2.z; b[11] = (_Float16)f2.w;
  b[12] = (_Float16)f3.x; b[13] = (_Float16)f3.y; b[14] = (_Float16)f3.z; b[15] = (_Float16)f3.w;
  return b;
}

__device__ __forceinline__ v16h load_a32_lds(const _Float16* __restrict__ row, int k, int lane) {
  const int klo = (lane & 16) ? 16 : 0;
  v16h a;
#pragma unroll
  for (int i = 0; i < 8; ++i) a[i] = row[k + klo + i];
#pragma unroll
  for (int i = 0; i < 8; ++i) a[8 + i] = row[k + klo + 8 + i];
  return a;
}

__device__ __forceinline__ v16h load_b32_strided(const float* __restrict__ base, int k,
                                                 int stride, int lane) {
  const int koff = (lane & 16) ? 16 : 0;
  v16h b;
#pragma unroll
  for (int i = 0; i < 16; ++i) b[i] = (_Float16)base[(size_t)(k + koff + i) * stride];
  return b;
}

// B fragment (32x16) from an LDS-resident W panel stored as Wl[n][k], k-stride 32
__device__ __forceinline__ v16h load_b_lds(const _Float16* __restrict__ Wl, int nt, int nlane,
                                           int koff) {
  const _Float16* __restrict__ r = Wl + (nt * 16 + nlane) * 32 + koff;
  v16h b;
#pragma unroll
  for (int i = 0; i < 16; ++i) b[i] = r[i];
  return b;
}

// ---------------------------------------------------------------------------
// Block-cooperative batched GEMM:
//   out[b][M,N] = act(scale * A[b][M,K] @ W[b][N,K]^T + bias)
// Block (256 thr = 8 waves) computes a 128x64 tile; the 64x32 W k-panel is
// staged once per k-step into LDS (f16) and shared by all waves. Each wave
// owns 16 rows x 64 cols (4 accumulators; A fragment reused across 4 WMMAs).
// M is clamped (not early-exited) so every wave reaches the block barriers;
// EXEC stays all-ones for WMMA.
// ---------------------------------------------------------------------------
__global__ void gemm_wmma_kernel(const float* __restrict__ A, const float* __restrict__ W,
                                 const float* __restrict__ bias, float* __restrict__ out,
                                 int M, int K, int N, long long aStride, long long wStride,
                                 long long oStride, int blocksN, int blocksPerBatch,
                                 float scale, int act) {
  __shared__ _Float16 Wl[64 * 32];
  const int tid   = threadIdx.x;
  const int wave  = tid >> 5;
  const int lane  = tid & 31;
  const int nlane = lane & 15;

  const int blk = blockIdx.x;
  const int b   = blk / blocksPerBatch;
  const int t   = blk - b * blocksPerBatch;
  const int bm  = t / blocksN;
  const int bn  = t - bm * blocksN;
  const int nbase = bn * 64;

  const int  mtile  = bm * 128 + wave * 16;
  const bool mvalid = (mtile < M);          // wave-uniform
  int mrow = mtile + nlane;
  if (mrow > M - 1) mrow = M - 1;           // clamp: keep loads legal, barriers reachable

  const float* __restrict__ Ab   = A + (long long)b * aStride;
  const float* __restrict__ Wb   = W + (long long)b * wStride;
  const float* __restrict__ arow = Ab + (size_t)mrow * K;

  // W staging coordinates: thread -> 8 contiguous k-values of one n-row
  const int sn = tid >> 2;          // 0..63
  const int sk = (tid & 3) * 8;     // 0,8,16,24
  const float* __restrict__ wstage = Wb + (size_t)(nbase + sn) * K + sk;
  _Float16* __restrict__ wdst = &Wl[sn * 32 + sk];

  v8f c0 = {}, c1 = {}, c2 = {}, c3 = {};
  const int koff = (lane & 16) ? 16 : 0;
  for (int k = 0; k < K; k += 32) {
    if (k) __syncthreads();  // protect previous iteration's LDS reads
    {
      const float4 f0 = *(const float4*)(wstage + k);
      const float4 f1 = *(const float4*)(wstage + k + 4);
      wdst[0] = (_Float16)f0.x; wdst[1] = (_Float16)f0.y;
      wdst[2] = (_Float16)f0.z; wdst[3] = (_Float16)f0.w;
      wdst[4] = (_Float16)f1.x; wdst[5] = (_Float16)f1.y;
      wdst[6] = (_Float16)f1.z; wdst[7] = (_Float16)f1.w;
    }
    __syncthreads();
    if (k + 32 < K) {  // prefetch next k-panels (global_prefetch_b8)
      __builtin_prefetch(arow + k + 32, 0, 0);
      __builtin_prefetch(wstage + k + 32, 0, 0);
    }
    const v16h af = load_a32(arow, k, lane);
    c0 = WMMA_F16(af, load_b_lds(Wl, 0, nlane, koff), c0);
    c1 = WMMA_F16(af, load_b_lds(Wl, 1, nlane, koff), c1);
    c2 = WMMA_F16(af, load_b_lds(Wl, 2, nlane, koff), c2);
    c3 = WMMA_F16(af, load_b_lds(Wl, 3, nlane, koff), c3);
  }

  if (mvalid) {
    float* __restrict__ Ob = out + (long long)b * oStride;
    const int r0 = mtile + ((lane & 16) ? 8 : 0);
    const v8f cc[4] = {c0, c1, c2, c3};
#pragma unroll
    for (int nt = 0; nt < 4; ++nt) {
      const int col = nbase + nt * 16 + nlane;
      const float bv = bias ? bias[col] : 0.0f;
#pragma unroll
      for (int r = 0; r < 8; ++r) {
        float v = cc[nt][r] * scale + bv;
        if (act == 1) v = 0.5f * v * (1.0f + erff(v * 0.70710678118654752f));  // exact GELU
        Ob[(size_t)(r0 + r) * N + col] = v;
      }
    }
  }
}

// ---------------------------------------------------------------------------
// Fused MHA: per (b,h) block, 8 waves x 16 query rows. Lq = Lk = 128, HD = 64.
// S = QK^T (WMMA) -> scale/mask/softmax (shfl within 16-lane half) -> P to LDS
// (f16, re-loaded in A-fragment layout) -> O = P V (WMMA, strided B frags).
// ---------------------------------------------------------------------------
#define PSTR 136
__global__ void attn_wmma_kernel(const float* __restrict__ Q, const float* __restrict__ Km,
                                 const float* __restrict__ V, float* __restrict__ O,
                                 int causal) {
  __shared__ _Float16 Pl[8 * 16 * PSTR];
  const int b    = blockIdx.x / H_;
  const int h    = blockIdx.x % H_;
  const int tid  = threadIdx.x;
  const int wave = tid >> 5;
  const int lane = tid & 31;
  const int qbase = wave * 16;
  const int nlane = lane & 15;
  const int roff  = (lane & 16) ? 8 : 0;

  const float* __restrict__ qrow = Q + ((size_t)(b * L_ + qbase + nlane)) * D_ + h * HD_;
  const v16h aq0 = load_a32(qrow, 0, lane);
  const v16h aq1 = load_a32(qrow, 32, lane);

  v8f S[8];
#pragma unroll
  for (int kt = 0; kt < 8; ++kt) {
    const float* __restrict__ krow = Km + ((size_t)(b * L_ + kt * 16 + nlane)) * D_ + h * HD_;
    v16h b0 = load_b32(krow, 0, lane);
    v16h b1 = load_b32(krow, 32, lane);
    v8f c = {};
    c = WMMA_F16(aq0, b0, c);
    c = WMMA_F16(aq1, b1, c);
    S[kt] = c;
  }

  // scale + optional causal mask
#pragma unroll
  for (int kt = 0; kt < 8; ++kt) {
#pragma unroll
    for (int r = 0; r < 8; ++r) {
      float v = S[kt][r] * 0.125f;  // 1/sqrt(64)
      if (causal && (kt * 16 + nlane) > (qbase + roff + r)) v = -__builtin_inff();
      S[kt][r] = v;
    }
  }

  // softmax per row (row spread across the lane's 16-lane half + 8 k-tiles)
#pragma unroll
  for (int r = 0; r < 8; ++r) {
    float mx = S[0][r];
#pragma unroll
    for (int kt = 1; kt < 8; ++kt) mx = fmaxf(mx, S[kt][r]);
    mx = fmaxf(mx, __shfl_xor(mx, 1, 32));
    mx = fmaxf(mx, __shfl_xor(mx, 2, 32));
    mx = fmaxf(mx, __shfl_xor(mx, 4, 32));
    mx = fmaxf(mx, __shfl_xor(mx, 8, 32));
    float sum = 0.0f;
#pragma unroll
    for (int kt = 0; kt < 8; ++kt) {
      float e = __expf(S[kt][r] - mx);
      S[kt][r] = e;
      sum += e;
    }
    sum += __shfl_xor(sum, 1, 32);
    sum += __shfl_xor(sum, 2, 32);
    sum += __shfl_xor(sum, 4, 32);
    sum += __shfl_xor(sum, 8, 32);
    const float inv = 1.0f / sum;
#pragma unroll
    for (int kt = 0; kt < 8; ++kt) S[kt][r] *= inv;
  }

  // stage P (f16) in this wave's LDS region
  _Float16* __restrict__ Pw = &Pl[wave * 16 * PSTR];
#pragma unroll
  for (int kt = 0; kt < 8; ++kt) {
#pragma unroll
    for (int r = 0; r < 8; ++r)
      Pw[(roff + r) * PSTR + kt * 16 + nlane] = (_Float16)S[kt][r];
  }
  __syncthreads();

  // O = P @ V
  const _Float16* __restrict__ prow = Pw + nlane * PSTR;
  const float* __restrict__ vbase   = V + (size_t)b * L_ * D_ + h * HD_;
#pragma unroll
  for (int nt = 0; nt < 4; ++nt) {
    v8f oc = {};
#pragma unroll
    for (int kt = 0; kt < 4; ++kt) {
      v16h ap = load_a32_lds(prow, kt * 32, lane);
      v16h bv = load_b32_strided(vbase + nt * 16 + nlane, kt * 32, D_, lane);
      oc = WMMA_F16(ap, bv, oc);
    }
#pragma unroll
    for (int r = 0; r < 8; ++r)
      O[((size_t)(b * L_ + qbase + roff + r)) * D_ + h * HD_ + nt * 16 + nlane] = oc[r];
  }
}

// ---------------------------------------------------------------------------
// Residual add + LayerNorm(D=512): Y = g * norm(X + R) + b. One row per block.
// ---------------------------------------------------------------------------
__global__ void add_ln_kernel(const float* __restrict__ X, const float* __restrict__ R,
                              const float* __restrict__ G, const float* __restrict__ Bt,
                              float* __restrict__ Y) {
  __shared__ float s1[256];
  __shared__ float s2[256];
  const size_t row = blockIdx.x;
  const int tid = threadIdx.x;
  const float* __restrict__ xr = X + row * D_;
  const float* __restrict__ rr = R + row * D_;
  const float v0 = xr[tid] + rr[tid];
  const float v1 = xr[tid + 256] + rr[tid + 256];
  s1[tid] = v0 + v1;
  s2[tid] = v0 * v0 + v1 * v1;
  __syncthreads();
  for (int s = 128; s > 0; s >>= 1) {
    if (tid < s) { s1[tid] += s1[tid + s]; s2[tid] += s2[tid + s]; }
    __syncthreads();
  }
  const float mean = s1[0] * (1.0f / D_);
  const float var  = s2[0] * (1.0f / D_) - mean * mean;
  const float inv  = rsqrtf(var + 1e-5f);
  Y[row * D_ + tid]       = (v0 - mean) * inv * G[tid] + Bt[tid];
  Y[row * D_ + tid + 256] = (v1 - mean) * inv * G[tid + 256] + Bt[tid + 256];
}

// decoder inputs: xfull[b,t,:] = enc[b,targets[b,t],:] + emb[t]; x0[b] = start + emb[0]
__global__ void build_dec_in_kernel(const float* __restrict__ enc, const int* __restrict__ tg,
                                    const float* __restrict__ emb, const float* __restrict__ st,
                                    float* __restrict__ xf, float* __restrict__ x0) {
  const int bt = blockIdx.x;
  const int b = bt / L_;
  const int t = bt % L_;
  const int sel = tg[bt];
  const int tid = threadIdx.x;
  const float* __restrict__ er = enc + ((size_t)b * L_ + sel) * D_;
  const float* __restrict__ em = emb + (size_t)t * D_;
  float* __restrict__ o = xf + (size_t)bt * D_;
  for (int i = tid; i < D_; i += 256) o[i] = er[i] + em[i];
  if (t == 0) {
    float* __restrict__ z = x0 + (size_t)b * D_;
    for (int i = tid; i < D_; i += 256) z[i] = st[i] + emb[i];
  }
}

// cross attention for the single start-token query (Lq = 1, Lk = 128)
__global__ void cross_attn_x0_kernel(const float* __restrict__ q0, const float* __restrict__ Kc,
                                     const float* __restrict__ Vc, float* __restrict__ o) {
  __shared__ float p[128];
  __shared__ float tmp[128];
  const int b = blockIdx.x / H_;
  const int h = blockIdx.x % H_;
  const int tid = threadIdx.x;  // key index
  const float* __restrict__ qv = q0 + (size_t)b * D_ + h * HD_;
  const float* __restrict__ kv = Kc + ((size_t)b * L_ + tid) * D_ + h * HD_;
  float s = 0.0f;
  for (int j = 0; j < HD_; ++j) s += qv[j] * kv[j];
  s *= 0.125f;
  tmp[tid] = s;
  __syncthreads();
  for (int st = 64; st > 0; st >>= 1) {
    if (tid < st) tmp[tid] = fmaxf(tmp[tid], tmp[tid + st]);
    __syncthreads();
  }
  const float mx = tmp[0];
  __syncthreads();
  const float e = __expf(s - mx);
  p[tid] = e;
  tmp[tid] = e;
  __syncthreads();
  for (int st = 64; st > 0; st >>= 1) {
    if (tid < st) tmp[tid] += tmp[tid + st];
    __syncthreads();
  }
  const float inv = 1.0f / tmp[0];
  if (tid < HD_) {
    float acc = 0.0f;
    for (int k2 = 0; k2 < L_; ++k2)
      acc += p[k2] * Vc[((size_t)b * L_ + k2) * D_ + h * HD_ + tid];
    o[(size_t)b * D_ + h * HD_ + tid] = acc * inv;
  }
}

// dec_out[b,0] = x0[b]; dec_out[b,t] = xfull[b,t-1]
__global__ void build_dec_out_kernel(const float* __restrict__ x0, const float* __restrict__ xf,
                                     float* __restrict__ dout) {
  const int bt = blockIdx.x;
  const int b = bt / L_;
  const int t = bt % L_;
  const int tid = threadIdx.x;
  const float* __restrict__ src =
      (t == 0) ? (x0 + (size_t)b * D_) : (xf + ((size_t)b * L_ + (t - 1)) * D_);
  float* __restrict__ o = dout + (size_t)bt * D_;
  for (int i = tid; i < D_; i += 256) o[i] = src[i];
}

__global__ void inv_perm_kernel(const int* __restrict__ tg, int* __restrict__ iv) {
  const int b = blockIdx.x;
  const int s = threadIdx.x;
  iv[b * L_ + tg[b * L_ + s]] = s;
}

__global__ void zero_one_kernel(float* p) {
  if (threadIdx.x == 0 && blockIdx.x == 0) *p = 0.0f;
}

// mask filled positions, softmax, log-prob at target, entropy accumulation
__global__ void final_softmax_kernel(const float* __restrict__ LG, const int* __restrict__ tg,
                                     const int* __restrict__ iv, float* __restrict__ operm,
                                     float* __restrict__ ologp, float* __restrict__ olog,
                                     float* __restrict__ oent) {
  __shared__ float tmp[128];
  __shared__ float pbuf[128];
  const int bt = blockIdx.x;
  const int b = bt / L_;
  const int t = bt % L_;
  const int l = threadIdx.x;
  float lg = LG[(size_t)bt * L_ + l];
  if (iv[b * L_ + l] < t) lg = -__builtin_inff();  // exclusive prefix fill mask
  olog[(size_t)bt * L_ + l] = lg;
  tmp[l] = lg;
  __syncthreads();
  for (int s = 64; s > 0; s >>= 1) {
    if (l < s) tmp[l] = fmaxf(tmp[l], tmp[l + s]);
    __syncthreads();
  }
  const float mx = tmp[0];
  __syncthreads();
  const float e = __expf(lg - mx);
  pbuf[l] = e;
  tmp[l] = e;
  __syncthreads();
  for (int s = 64; s > 0; s >>= 1) {
    if (l < s) tmp[l] += tmp[l + s];
    __syncthreads();
  }
  const float inv = 1.0f / tmp[0];
  __syncthreads();
  const float pr = pbuf[l] * inv;
  const float plogp = (pr > 0.0f) ? pr * logf(fmaxf(pr, 1e-38f)) : 0.0f;
  tmp[l] = plogp;
  __syncthreads();
  for (int s = 64; s > 0; s >>= 1) {
    if (l < s) tmp[l] += tmp[l + s];
    __syncthreads();
  }
  if (l == 0) {
    atomicAdd(oent, (-tmp[0]) / (float)(B_ * L_));
    const int tt = tg[bt];
    ologp[bt] = logf(pbuf[tt] * inv + 1e-10f);
    operm[bt] = (float)tt;
  }
}

// ---------------------------------------------------------------------------
// Host orchestration
// ---------------------------------------------------------------------------
extern "C" void kernel_launch(void* const* d_in, const int* in_sizes, int n_in,
                              void* d_out, int out_size, void* d_ws, size_t ws_size,
                              hipStream_t stream) {
  (void)in_sizes; (void)n_in; (void)out_size; (void)ws_size;

  // jax pytree leaf order: dict keys sorted alphabetically, depth-first.
  const float* vectors   = (const float*)d_in[0];   // [B,L,DV]
  const int*   targets   = (const int*)d_in[1];     // [B,L]
  const float* dec_b1    = (const float*)d_in[2];   // [4,2048]
  const float* dec_b2    = (const float*)d_in[3];   // [4,512]
  const float* dec_ca_ob = (const float*)d_in[4];   // [4,512]
  const float* dec_ca_ow = (const float*)d_in[5];   // [4,512,512]
  const float* dec_ca_qb = (const float*)d_in[6];   // [4,1536]
  const float* dec_ca_qw = (const float*)d_in[7];   // [4,1536,512]
  const float* dec_n1b   = (const float*)d_in[8];
  const float* dec_n1g   = (const float*)d_in[9];
  const float* dec_n2b   = (const float*)d_in[10];
  const float* dec_n2g   = (const float*)d_in[11];
  const float* dec_n3b   = (const float*)d_in[12];
  const float* dec_n3g   = (const float*)d_in[13];
  const float* dec_sa_ob = (const float*)d_in[14];
  const float* dec_sa_ow = (const float*)d_in[15];
  const float* dec_sa_qb = (const float*)d_in[16];
  const float* dec_sa_qw = (const float*)d_in[17];
  const float* dec_w1    = (const float*)d_in[18];  // [4,2048,512]
  const float* dec_w2    = (const float*)d_in[19];  // [4,512,2048]
  const float* enc_in_b  = (const float*)d_in[20];  // [512]
  const float* enc_in_w  = (const float*)d_in[21];  // [512,64]
  const float* enc_ob    = (const float*)d_in[22];
  const float* enc_ow    = (const float*)d_in[23];
  const float* enc_qb    = (const float*)d_in[24];
  const float* enc_qw    = (const float*)d_in[25];
  const float* enc_b1    = (const float*)d_in[26];
  const float* enc_b2    = (const float*)d_in[27];
  const float* enc_n1b   = (const float*)d_in[28];
  const float* enc_n1g   = (const float*)d_in[29];
  const float* enc_n2b   = (const float*)d_in[30];
  const float* enc_n2g   = (const float*)d_in[31];
  const float* enc_w1    = (const float*)d_in[32];
  const float* enc_w2    = (const float*)d_in[33];
  const float* k_b       = (const float*)d_in[34];
  const float* k_w       = (const float*)d_in[35];
  const float* q_b       = (const float*)d_in[36];
  const float* q_w       = (const float*)d_in[37];
  const float* start_tok = (const float*)d_in[38];  // [1,1,512]
  const float* step_emb  = (const float*)d_in[39];  // [128,512]

  // workspace carving (floats)
  float* wsf = (float*)d_ws;
  size_t off = 0;
  auto carve = [&](size_t n) { float* p = wsf + off; off += n; return p; };
  float* E   = carve((size_t)BL_ * D_);   // encoder output (persists)
  float* P0  = carve((size_t)BL_ * D_);   // x_full ping
  float* P1  = carve((size_t)BL_ * D_);   // x_full pong
  float* KC  = carve((size_t)BL_ * D_);   // cross-attn K (per layer)
  float* VC  = carve((size_t)BL_ * D_);   // cross-attn V (per layer)
  float* QB  = carve((size_t)BL_ * D_);
  float* KB  = carve((size_t)BL_ * D_);
  float* VB  = carve((size_t)BL_ * D_);
  float* AO  = carve((size_t)BL_ * D_);
  float* PR  = carve((size_t)BL_ * D_);
  float* FFb = carve((size_t)BL_ * FF_);
  float* F2  = carve((size_t)BL_ * D_);
  float* Z0  = carve((size_t)B_ * D_);    // x0 ping
  float* Z1  = carve((size_t)B_ * D_);    // x0 pong
  float* ZQ  = carve((size_t)B_ * D_);
  float* ZA  = carve((size_t)B_ * D_);
  float* ZP  = carve((size_t)B_ * D_);
  float* ZF  = carve((size_t)B_ * FF_);
  int*   IV  = (int*)carve((size_t)BL_);
  float* DOb = AO;   // reuse at readout stage
  float* QF  = QB;
  float* KF  = KB;
  float* LG  = FFb;  // [B,L,L] fits in the FF buffer

  auto gemm = [&](const float* A, const float* W, const float* bias, float* out,
                  int M, int N, int K, int batch, long long as, long long ws2,
                  long long os, float scale, int act) {
    const int bM  = (M + 127) / 128;
    const int bN  = N / 64;
    const int bpb = bM * bN;
    gemm_wmma_kernel<<<dim3(batch * bpb), dim3(256), 0, stream>>>(
        A, W, bias, out, M, K, N, as, ws2, os, bN, bpb, scale, act);
  };
  auto gemmNB = [&](const float* A, const float* W, const float* bias, float* out,
                    int M, int N, int K, int act) {
    gemm(A, W, bias, out, M, N, K, 1, 0, 0, 0, 1.0f, act);
  };

  // ---------------- encoder ----------------
  gemmNB(vectors, enc_in_w, enc_in_b, E, BL_, D_, DV_, 0);
  for (int j = 0; j < NE_; ++j) {
    const float* qw  = enc_qw + (size_t)j * 3 * D_ * D_;
    const float* qb2 = enc_qb + (size_t)j * 3 * D_;
    gemmNB(E, qw,                qb2,          QB, BL_, D_, D_, 0);
    gemmNB(E, qw + D_ * D_,      qb2 + D_,     KB, BL_, D_, D_, 0);
    gemmNB(E, qw + 2 * D_ * D_,  qb2 + 2 * D_, VB, BL_, D_, D_, 0);
    attn_wmma_kernel<<<B_ * H_, 256, 0, stream>>>(QB, KB, VB, AO, 0);
    gemmNB(AO, enc_ow + (size_t)j * D_ * D_, enc_ob + (size_t)j * D_, PR, BL_, D_, D_, 0);
    add_ln_kernel<<<BL_, 256, 0, stream>>>(E, PR, enc_n1g + (size_t)j * D_,
                                           enc_n1b + (size_t)j * D_, P0);
    gemmNB(P0, enc_w1 + (size_t)j * FF_ * D_, enc_b1 + (size_t)j * FF_, FFb, BL_, FF_, D_, 1);
    gemmNB(FFb, enc_w2 + (size_t)j * D_ * FF_, enc_b2 + (size_t)j * D_, F2, BL_, D_, FF_, 0);
    add_ln_kernel<<<BL_, 256, 0, stream>>>(P0, F2, enc_n2g + (size_t)j * D_,
                                           enc_n2b + (size_t)j * D_, E);
  }

  // ---------------- decoder inputs ----------------
  build_dec_in_kernel<<<BL_, 256, 0, stream>>>(E, targets, step_emb, start_tok, P0, Z0);

  // ---------------- decoder ----------------
  float* Xf = P0; float* Xf2 = P1;
  float* z0 = Z0; float* z1 = Z1;
  for (int j = 0; j < ND_; ++j) {
    const float* saw  = dec_sa_qw + (size_t)j * 3 * D_ * D_;
    const float* sab  = dec_sa_qb + (size_t)j * 3 * D_;
    const float* saow = dec_sa_ow + (size_t)j * D_ * D_;
    const float* saob = dec_sa_ob + (size_t)j * D_;
    const float* caw  = dec_ca_qw + (size_t)j * 3 * D_ * D_;
    const float* cab  = dec_ca_qb + (size_t)j * 3 * D_;
    const float* caow = dec_ca_ow + (size_t)j * D_ * D_;
    const float* caob = dec_ca_ob + (size_t)j * D_;
    const float* w1   = dec_w1 + (size_t)j * FF_ * D_;
    const float* b1   = dec_b1 + (size_t)j * FF_;
    const float* w2   = dec_w2 + (size_t)j * D_ * FF_;
    const float* b2   = dec_b2 + (size_t)j * D_;

    // x_full: causal self-attention
    gemmNB(Xf, saw,               sab,          QB, BL_, D_, D_, 0);
    gemmNB(Xf, saw + D_ * D_,     sab + D_,     KB, BL_, D_, D_, 0);
    gemmNB(Xf, saw + 2 * D_ * D_, sab + 2 * D_, VB, BL_, D_, D_, 0);
    attn_wmma_kernel<<<B_ * H_, 256, 0, stream>>>(QB, KB, VB, AO, 1);
    gemmNB(AO, saow, saob, PR, BL_, D_, D_, 0);
    add_ln_kernel<<<BL_, 256, 0, stream>>>(Xf, PR, dec_n1g + (size_t)j * D_,
                                           dec_n1b + (size_t)j * D_, Xf2);
    // x_full: cross-attention (K/V from encoder output; shared with x0 pass)
    gemmNB(Xf2, caw,               cab,          QB, BL_, D_, D_, 0);
    gemmNB(E,   caw + D_ * D_,     cab + D_,     KC, BL_, D_, D_, 0);
    gemmNB(E,   caw + 2 * D_ * D_, cab + 2 * D_, VC, BL_, D_, D_, 0);
    attn_wmma_kernel<<<B_ * H_, 256, 0, stream>>>(QB, KC, VC, AO, 0);
    gemmNB(AO, caow, caob, PR, BL_, D_, D_, 0);
    add_ln_kernel<<<BL_, 256, 0, stream>>>(Xf2, PR, dec_n2g + (size_t)j * D_,
                                           dec_n2b + (size_t)j * D_, Xf);
    // x_full: FFN
    gemmNB(Xf, w1, b1, FFb, BL_, FF_, D_, 1);
    gemmNB(FFb, w2, b2, F2, BL_, D_, FF_, 0);
    add_ln_kernel<<<BL_, 256, 0, stream>>>(Xf, F2, dec_n3g + (size_t)j * D_,
                                           dec_n3b + (size_t)j * D_, Xf2);

    // x0 path: T=1 self-attn => softmax over a single key => o = v-projection
    gemmNB(z0, saw + 2 * D_ * D_, sab + 2 * D_, ZQ, B_, D_, D_, 0);
    gemmNB(ZQ, saow, saob, ZP, B_, D_, D_, 0);
    add_ln_kernel<<<B_, 256, 0, stream>>>(z0, ZP, dec_n1g + (size_t)j * D_,
                                          dec_n1b + (size_t)j * D_, z1);
    // x0 cross-attention (reuses KC/VC)
    gemmNB(z1, caw, cab, ZQ, B_, D_, D_, 0);
    cross_attn_x0_kernel<<<B_ * H_, 128, 0, stream>>>(ZQ, KC, VC, ZA);
    gemmNB(ZA, caow, caob, ZP, B_, D_, D_, 0);
    add_ln_kernel<<<B_, 256, 0, stream>>>(z1, ZP, dec_n2g + (size_t)j * D_,
                                          dec_n2b + (size_t)j * D_, z0);
    // x0 FFN
    gemmNB(z0, w1, b1, ZF, B_, FF_, D_, 1);
    gemmNB(ZF, w2, b2, ZP, B_, D_, FF_, 0);
    add_ln_kernel<<<B_, 256, 0, stream>>>(z0, ZP, dec_n3g + (size_t)j * D_,
                                          dec_n3b + (size_t)j * D_, z1);

    float* t_ = Xf; Xf = Xf2; Xf2 = t_;
    t_ = z0; z0 = z1; z1 = t_;
  }

  // ---------------- pointer-network readout ----------------
  build_dec_out_kernel<<<BL_, 256, 0, stream>>>(z0, Xf, DOb);
  gemmNB(DOb, q_w, q_b, QF, BL_, D_, D_, 0);
  gemmNB(E,   k_w, k_b, KF, BL_, D_, D_, 0);
  gemm(QF, KF, nullptr, LG, L_, L_, D_, B_,
       (long long)L_ * D_, (long long)L_ * D_, (long long)L_ * L_,
       1.0f / sqrtf((float)D_), 0);
  inv_perm_kernel<<<B_, 128, 0, stream>>>(targets, IV);

  float* out   = (float*)d_out;
  float* operm = out;
  float* ologp = out + BL_;
  float* olog  = out + 2 * BL_;
  float* oent  = out + 2 * BL_ + (size_t)B_ * L_ * L_;
  zero_one_kernel<<<1, 1, 0, stream>>>(oent);
  final_softmax_kernel<<<BL_, 128, 0, stream>>>(LG, targets, IV, operm, ologp, olog, oent);
}
